// QuantumLayer_9483287790321
// MI455X (gfx1250) — matrix operand order repaired and checked by reading, not compiled, not run on hardware
//
#include <hip/hip_runtime.h>
#include <cstdint>

// ---------------------------------------------------------------------------
// QuantumLayer fused kernel for gfx1250 (MI455X, wave32, WMMA).
//   B=8, L=2048, D_MODEL=512, N_QUBITS=10, Q_LAYERS=4, DIM=1024
//   One 256-thread block = 8 wave32 = 8 tokens. State vector (1024 c64) held
//   as re[32]/im[32] f32 VGPRs per lane. Lane-crossing qubit gates use
//   ds_swizzle_b32 SWAPX patterns. Both GEMMs use v_wmma_f32_16x16x32_f16.
//   v2: branchless + vectorized (b128/b64) WMMA fragment loads, LDS-staged
//   vectorized epilogue (removes the saveexec/predicated-load chains).
// ---------------------------------------------------------------------------

typedef __attribute__((ext_vector_type(16))) _Float16 v16h;
typedef __attribute__((ext_vector_type(8)))  float    v8f;

#define QN          10
#define QLAYERS     4
#define DM          512
#define RPL         32      // complex amplitudes per lane (1024 / 32)
#define TOK_PER_BLK 8
#define NTOK        (8 * 2048)

// ds_swizzle group-of-32 XOR exchange: offset = {xor<<10 | or<<5 | and}
template<int XM>
__device__ __forceinline__ float swz(float v) {
  int i = __builtin_amdgcn_ds_swizzle(__float_as_int(v), (XM << 10) | 0x1f);
  return __int_as_float(i);
}

// ---- RX(theta): [[c, -i s], [-i s, c]] ; symmetric across the pair --------
template<int LM>  // lane-bit qubit
__device__ __forceinline__ void rx_lane(float (&re)[RPL], float (&im)[RPL],
                                        float c, float s) {
#pragma unroll
  for (int r = 0; r < RPL; ++r) {
    float pr = swz<LM>(re[r]);
    float pi = swz<LM>(im[r]);
    float nr = fmaf(s, pi, c * re[r]);
    float ni = fmaf(-s, pr, c * im[r]);
    re[r] = nr; im[r] = ni;
  }
}

template<int RM>  // register-bit qubit
__device__ __forceinline__ void rx_reg(float (&re)[RPL], float (&im)[RPL],
                                       float c, float s) {
#pragma unroll
  for (int r = 0; r < RPL; ++r) {
    if ((r & RM) == 0) {
      const int p = r | RM;
      float r0 = re[r], i0 = im[r], r1 = re[p], i1 = im[p];
      re[r] = fmaf(s, i1, c * r0);
      im[r] = fmaf(-s, r1, c * i0);
      re[p] = fmaf(s, i0, c * r1);
      im[p] = fmaf(-s, r0, c * i1);
    }
  }
}

// ---- phase diag(c - i s, c + i s): amp *= (c + i*seff), seff = +-s --------
__device__ __forceinline__ void ph_lane(float (&re)[RPL], float (&im)[RPL],
                                        float c, float seff) {
#pragma unroll
  for (int r = 0; r < RPL; ++r) {
    float nr = fmaf(-seff, im[r], c * re[r]);
    float ni = fmaf(seff, re[r], c * im[r]);
    re[r] = nr; im[r] = ni;
  }
}

template<int RM>
__device__ __forceinline__ void ph_reg(float (&re)[RPL], float (&im)[RPL],
                                       float c, float s) {
#pragma unroll
  for (int r = 0; r < RPL; ++r) {
    const float se = (r & RM) ? s : -s;
    float nr = fmaf(-se, im[r], c * re[r]);
    float ni = fmaf(se, re[r], c * im[r]);
    re[r] = nr; im[r] = ni;
  }
}

// ---- RY(theta): [[c,-s],[s,c]] (real rotation on re and im alike) --------
template<int LM>
__device__ __forceinline__ void ry_lane(float (&re)[RPL], float (&im)[RPL],
                                        float c, float seff) {
#pragma unroll
  for (int r = 0; r < RPL; ++r) {
    float pr = swz<LM>(re[r]);
    float pi = swz<LM>(im[r]);
    re[r] = fmaf(seff, pr, c * re[r]);
    im[r] = fmaf(seff, pi, c * im[r]);
  }
}

template<int RM>
__device__ __forceinline__ void ry_reg(float (&re)[RPL], float (&im)[RPL],
                                       float c, float s) {
#pragma unroll
  for (int r = 0; r < RPL; ++r) {
    if ((r & RM) == 0) {
      const int p = r | RM;
      float r0 = re[r], r1 = re[p], i0 = im[r], i1 = im[p];
      re[r] = fmaf(-s, r1, c * r0);
      re[p] = fmaf(s, r0, c * r1);
      im[r] = fmaf(-s, i1, c * i0);
      im[p] = fmaf(s, i0, c * i1);
    }
  }
}

// ---- CNOT variants --------------------------------------------------------
template<int CM, int TM>  // control & target both register bits
__device__ __forceinline__ void cnot_rr(float (&re)[RPL], float (&im)[RPL]) {
#pragma unroll
  for (int r = 0; r < RPL; ++r) {
    if ((r & CM) && !(r & TM)) {
      const int p = r | TM;
      float t = re[r]; re[r] = re[p]; re[p] = t;
      t = im[r]; im[r] = im[p]; im[p] = t;
    }
  }
}

template<int CM, int LM>  // control register bit, target lane bit
__device__ __forceinline__ void cnot_rl(float (&re)[RPL], float (&im)[RPL]) {
#pragma unroll
  for (int r = 0; r < RPL; ++r) {
    if (r & CM) {
      re[r] = swz<LM>(re[r]);
      im[r] = swz<LM>(im[r]);
    }
  }
}

template<int CB, int TM>  // control lane bit CB, target lane mask TM
__device__ __forceinline__ void cnot_ll(float (&re)[RPL], float (&im)[RPL],
                                        int lane) {
  const bool cb = (lane >> CB) & 1;
#pragma unroll
  for (int r = 0; r < RPL; ++r) {
    float pr = swz<TM>(re[r]);
    float pi = swz<TM>(im[r]);
    re[r] = cb ? pr : re[r];
    im[r] = cb ? pi : im[r];
  }
}

template<int CB, int TM>  // control lane bit CB, target register mask TM
__device__ __forceinline__ void cnot_lr(float (&re)[RPL], float (&im)[RPL],
                                        int lane) {
  const bool cb = (lane >> CB) & 1;
#pragma unroll
  for (int r = 0; r < RPL; ++r) {
    if (!(r & TM)) {
      const int p = r | TM;
      float a0 = re[r], a1 = re[p];
      re[r] = cb ? a1 : a0; re[p] = cb ? a0 : a1;
      a0 = im[r]; a1 = im[p];
      im[r] = cb ? a1 : a0; im[p] = cb ? a0 : a1;
    }
  }
}

// ---------------------------------------------------------------------------
__global__ void __launch_bounds__(256, 1)
quantum_layer_kernel(const float* __restrict__ x,
                     const float* __restrict__ in_W,
                     const float* __restrict__ in_b,
                     const float* __restrict__ gamma,
                     const float* __restrict__ beta,
                     const float* __restrict__ weights,
                     const float* __restrict__ out_W,
                     const float* __restrict__ out_b,
                     float* __restrict__ out) {
  __shared__ float sh_h[16][16];             // stage-1 WMMA D tile
  __shared__ float sh_f[TOK_PER_BLK][QN];    // layer-normed features
  __shared__ float sh_z[TOK_PER_BLK][QN];    // <Z_q> per token
  __shared__ float sh_o[TOK_PER_BLK][DM];    // stage-3 GEMM result tile

  const int tid     = threadIdx.x;
  const int lane    = tid & 31;
  const int wave    = tid >> 5;
  const int tokBase = blockIdx.x * TOK_PER_BLK;

  const int mrow  = lane & 15;   // WMMA row/col index within half
  const int khalf = lane >> 4;   // which 16-lane half

  // Warm L2/WGP$ for this block's x rows (global_prefetch_b8).
  __builtin_prefetch(x + (size_t)(tokBase + wave) * DM + lane * 16, 0, 1);

  // ================= Stage 1: h = x @ in_W^T via WMMA (wave 0) =============
  if (wave == 0) {
    v8f acc = {0.f, 0.f, 0.f, 0.f, 0.f, 0.f, 0.f, 0.f};
    const int tok   = tokBase + (mrow & 7);        // rows 8..15 dup 0..7
    const int wrow  = (mrow < QN) ? mrow : 0;      // clamped -> loads in-bounds
    const bool bval = (mrow < QN);
#pragma unroll
    for (int kt = 0; kt < DM / 32; ++kt) {
      const int k0 = kt * 32;
      v16h a, bfrag;
      // A 16x32 f16 layout: per lane two groups of 8 consecutive K values.
#pragma unroll
      for (int g = 0; g < 2; ++g) {
        const float* xp = x + (size_t)tok * DM + k0 + khalf * 16 + g * 8;
        const float4 x0 = *(const float4*)(xp);
        const float4 x1 = *(const float4*)(xp + 4);
        a[g * 8 + 0] = (_Float16)x0.x; a[g * 8 + 1] = (_Float16)x0.y;
        a[g * 8 + 2] = (_Float16)x0.z; a[g * 8 + 3] = (_Float16)x0.w;
        a[g * 8 + 4] = (_Float16)x1.x; a[g * 8 + 5] = (_Float16)x1.y;
        a[g * 8 + 6] = (_Float16)x1.z; a[g * 8 + 7] = (_Float16)x1.w;
      }
      // B 32x16 f16 layout: 16 consecutive K values per lane.
      const float* wp = in_W + wrow * DM + k0 + khalf * 16;
      const float4 w0 = *(const float4*)(wp);
      const float4 w1 = *(const float4*)(wp + 4);
      const float4 w2 = *(const float4*)(wp + 8);
      const float4 w3 = *(const float4*)(wp + 12);
      const float wv[16] = {w0.x, w0.y, w0.z, w0.w, w1.x, w1.y, w1.z, w1.w,
                            w2.x, w2.y, w2.z, w2.w, w3.x, w3.y, w3.z, w3.w};
#pragma unroll
      for (int e = 0; e < 16; ++e)
        bfrag[e] = bval ? (_Float16)wv[e] : (_Float16)0.f;  // cndmask, no branch
      acc = __builtin_amdgcn_wmma_f32_16x16x32_f16(
          false, a, false, bfrag, (short)0, acc, false, false);
    }
#pragma unroll
    for (int v = 0; v < 8; ++v)
      sh_h[khalf * 8 + v][mrow] = acc[v];   // D: M = khalf*8+v, N = mrow

    // tanh + LayerNorm over the qubit dim, one token per lane.
    if (lane < TOK_PER_BLK) {
      float hb[QN];
      float mu = 0.f;
#pragma unroll
      for (int q = 0; q < QN; ++q) {
        hb[q] = tanhf(sh_h[lane][q] + in_b[q]);
        mu += hb[q];
      }
      mu *= (1.0f / QN);
      float var = 0.f;
#pragma unroll
      for (int q = 0; q < QN; ++q) {
        const float d = hb[q] - mu;
        var = fmaf(d, d, var);
      }
      var *= (1.0f / QN);
      const float inv = rsqrtf(var + 1e-5f);
#pragma unroll
      for (int q = 0; q < QN; ++q)
        sh_f[lane][q] = (hb[q] - mu) * inv * gamma[q] + beta[q];
    }
  }
  __syncthreads();

  // ================= Stage 2: 10-qubit statevector sim, 1 wave = 1 token ===
  // index i = r*32 + lane ; qubit q <-> index bit (9-q)
  float cx[QN], sx[QN];
#pragma unroll
  for (int q = 0; q < QN; ++q) {
    const float ha = 0.5f * sh_f[wave][q];
    cx[q] = cosf(ha);
    sx[q] = sinf(ha);
  }
  float bsg[5];  // +1 if lane bit set else -1
#pragma unroll
  for (int b = 0; b < 5; ++b)
    bsg[b] = ((lane >> b) & 1) ? 1.f : -1.f;

  float re[RPL], im[RPL];
#pragma unroll
  for (int r = 0; r < RPL; ++r) { re[r] = 0.f; im[r] = 0.f; }
  re[0] = (lane == 0) ? 1.f : 0.f;

#pragma unroll 1
  for (int l = 0; l < QLAYERS; ++l) {
    // RX(feature) on all qubits
    rx_reg<16>(re, im, cx[0], sx[0]);
    rx_reg< 8>(re, im, cx[1], sx[1]);
    rx_reg< 4>(re, im, cx[2], sx[2]);
    rx_reg< 2>(re, im, cx[3], sx[3]);
    rx_reg< 1>(re, im, cx[4], sx[4]);
    rx_lane<16>(re, im, cx[5], sx[5]);
    rx_lane< 8>(re, im, cx[6], sx[6]);
    rx_lane< 4>(re, im, cx[7], sx[7]);
    rx_lane< 2>(re, im, cx[8], sx[8]);
    rx_lane< 1>(re, im, cx[9], sx[9]);
    // phase(feature): phz = cos + i sin of the same half-angle
    ph_reg<16>(re, im, cx[0], sx[0]);
    ph_reg< 8>(re, im, cx[1], sx[1]);
    ph_reg< 4>(re, im, cx[2], sx[2]);
    ph_reg< 2>(re, im, cx[3], sx[3]);
    ph_reg< 1>(re, im, cx[4], sx[4]);
    ph_lane(re, im, cx[5], sx[5] * bsg[4]);
    ph_lane(re, im, cx[6], sx[6] * bsg[3]);
    ph_lane(re, im, cx[7], sx[7] * bsg[2]);
    ph_lane(re, im, cx[8], sx[8] * bsg[1]);
    ph_lane(re, im, cx[9], sx[9] * bsg[0]);
    // trainable RY + RZ
    float cy[QN], sy[QN], cz[QN], szv[QN];
#pragma unroll
    for (int q = 0; q < QN; ++q) {
      const float ty = 0.5f * weights[(l * QN + q) * 2 + 0];
      const float tz = 0.5f * weights[(l * QN + q) * 2 + 1];
      cy[q] = cosf(ty); sy[q]  = sinf(ty);
      cz[q] = cosf(tz); szv[q] = sinf(tz);
    }
    ry_reg<16>(re, im, cy[0], sy[0]);  ph_reg<16>(re, im, cz[0], szv[0]);
    ry_reg< 8>(re, im, cy[1], sy[1]);  ph_reg< 8>(re, im, cz[1], szv[1]);
    ry_reg< 4>(re, im, cy[2], sy[2]);  ph_reg< 4>(re, im, cz[2], szv[2]);
    ry_reg< 2>(re, im, cy[3], sy[3]);  ph_reg< 2>(re, im, cz[3], szv[3]);
    ry_reg< 1>(re, im, cy[4], sy[4]);  ph_reg< 1>(re, im, cz[4], szv[4]);
    ry_lane<16>(re, im, cy[5], sy[5] * bsg[4]); ph_lane(re, im, cz[5], szv[5] * bsg[4]);
    ry_lane< 8>(re, im, cy[6], sy[6] * bsg[3]); ph_lane(re, im, cz[6], szv[6] * bsg[3]);
    ry_lane< 4>(re, im, cy[7], sy[7] * bsg[2]); ph_lane(re, im, cz[7], szv[7] * bsg[2]);
    ry_lane< 2>(re, im, cy[8], sy[8] * bsg[1]); ph_lane(re, im, cz[8], szv[8] * bsg[1]);
    ry_lane< 1>(re, im, cy[9], sy[9] * bsg[0]); ph_lane(re, im, cz[9], szv[9] * bsg[0]);
    // CNOT ring: (0,1)..(8,9),(9,0)
    cnot_rr<16, 8>(re, im);
    cnot_rr< 8, 4>(re, im);
    cnot_rr< 4, 2>(re, im);
    cnot_rr< 2, 1>(re, im);
    cnot_rl< 1, 16>(re, im);
    cnot_ll<4, 8>(re, im, lane);
    cnot_ll<3, 4>(re, im, lane);
    cnot_ll<2, 2>(re, im, lane);
    cnot_ll<1, 1>(re, im, lane);
    cnot_lr<0, 16>(re, im, lane);
  }

  // Measurement: z_q = sum_i p_i * (bit_q(i) ? -1 : +1)
  float zq[QN];
#pragma unroll
  for (int q = 0; q < QN; ++q) zq[q] = 0.f;
  float ptot = 0.f;
#pragma unroll
  for (int r = 0; r < RPL; ++r) {
    const float p = fmaf(re[r], re[r], im[r] * im[r]);
    zq[0] += (r & 16) ? -p : p;
    zq[1] += (r &  8) ? -p : p;
    zq[2] += (r &  4) ? -p : p;
    zq[3] += (r &  2) ? -p : p;
    zq[4] += (r &  1) ? -p : p;
    ptot  += p;
  }
  zq[5] = -bsg[4] * ptot;
  zq[6] = -bsg[3] * ptot;
  zq[7] = -bsg[2] * ptot;
  zq[8] = -bsg[1] * ptot;
  zq[9] = -bsg[0] * ptot;

#pragma unroll
  for (int q = 0; q < QN; ++q) {  // 5-step butterfly wave reduction
    float v = zq[q];
    v += swz<1>(v);
    v += swz<2>(v);
    v += swz<4>(v);
    v += swz<8>(v);
    v += swz<16>(v);
    zq[q] = v;
  }
  if (lane == 0) {
#pragma unroll
    for (int q = 0; q < QN; ++q) sh_z[wave][q] = zq[q];
  }
  __syncthreads();

  // ========= Stage 3: z @ out_W^T via WMMA, staged to LDS ==================
  v16h afrag;
#pragma unroll
  for (int e = 0; e < 16; ++e) {
    const int pr  = e >> 1;
    const int Ka  = (khalf << 4) + e;
    const int KaC = (Ka < QN) ? Ka : 0;                    // clamped LDS read
    const float v = sh_z[mrow & 7][KaC];
    afrag[e] = (Ka < QN) ? (_Float16)v : (_Float16)0.f;    // cndmask
  }
#pragma unroll
  for (int sub = 0; sub < 4; ++sub) {
    const int n0  = wave * 64 + sub * 16;  // 8 waves x 64 cols = 512
    const int col = n0 + mrow;
    // out_W row = 10 consecutive floats (8B aligned) -> 5x b64 loads
    const float* op = out_W + col * QN;
    float wv[QN];
#pragma unroll
    for (int j = 0; j < 5; ++j) {
      const float2 w2 = *(const float2*)(op + 2 * j);
      wv[2 * j] = w2.x; wv[2 * j + 1] = w2.y;
    }
    v16h bfrag;
#pragma unroll
    for (int e = 0; e < 16; ++e) {
      if (e < QN)
        bfrag[e] = (khalf == 0) ? (_Float16)wv[e] : (_Float16)0.f;
      else
        bfrag[e] = (_Float16)0.f;
    }
    v8f dacc = {0.f, 0.f, 0.f, 0.f, 0.f, 0.f, 0.f, 0.f};
    dacc = __builtin_amdgcn_wmma_f32_16x16x32_f16(
        false, afrag, false, bfrag, (short)0, dacc, false, false);
    if (khalf == 0) {               // rows M=0..7 are the valid tokens
#pragma unroll
      for (int v = 0; v < 8; ++v) sh_o[v][col] = dacc[v];
    }
  }
  __syncthreads();

  // ========= Epilogue: out = x + sh_o + out_b, fully vectorized b128 =======
#pragma unroll
  for (int j = 0; j < 4; ++j) {
    const int off = tid * 16 + j * 4;        // 256 thr * 16 = 4096 = 8*512
    const int tk  = off >> 9;
    const int cl  = off & (DM - 1);
    const size_t gidx = (size_t)(tokBase + tk) * DM + cl;
    const float4 xv = *(const float4*)(x + gidx);
    const float4 bb = *(const float4*)(out_b + cl);
    float4 o;
    o.x = xv.x + bb.x + sh_o[tk][cl + 0];
    o.y = xv.y + bb.y + sh_o[tk][cl + 1];
    o.z = xv.z + bb.z + sh_o[tk][cl + 2];
    o.w = xv.w + bb.w + sh_o[tk][cl + 3];
    *(float4*)(out + gidx) = o;
  }
}

// ---------------------------------------------------------------------------
extern "C" void kernel_launch(void* const* d_in, const int* in_sizes, int n_in,
                              void* d_out, int out_size, void* d_ws,
                              size_t ws_size, hipStream_t stream) {
  (void)in_sizes; (void)n_in; (void)out_size; (void)d_ws; (void)ws_size;
  const float* x       = (const float*)d_in[0];
  const float* in_W    = (const float*)d_in[1];
  const float* in_b    = (const float*)d_in[2];
  const float* gamma   = (const float*)d_in[3];
  const float* beta    = (const float*)d_in[4];
  const float* weights = (const float*)d_in[5];
  const float* out_W   = (const float*)d_in[6];
  const float* out_b   = (const float*)d_in[7];
  float* out = (float*)d_out;

  dim3 grid(NTOK / TOK_PER_BLK);   // 2048 blocks
  dim3 block(256);                 // 8 wave32 = 8 tokens per block
  quantum_layer_kernel<<<grid, block, 0, stream>>>(
      x, in_W, in_b, gamma, beta, weights, out_W, out_b, out);
}
